// LAMOI_27230092657500
// MI455X (gfx1250) — matrix operand order repaired and checked by reading, not compiled, not run on hardware
//
#include <hip/hip_runtime.h>

typedef __attribute__((ext_vector_type(8)))  __bf16 v8bf;
typedef __attribute__((ext_vector_type(16))) __bf16 v16bf;
typedef __attribute__((ext_vector_type(8)))  float  v8f;
typedef __attribute__((ext_vector_type(4)))  unsigned int u32x4;
typedef __attribute__((ext_vector_type(4)))  int i32x4;
typedef __attribute__((ext_vector_type(8)))  int i32x8;

union FragU { v16bf v; v8bf h[2]; };
union AccU  { v8f v; float f[8]; };

// ---------------------------------------------------------------------------
// TDM: DMA one 128-row x 32-col bf16 tile (row stride = strideElems) from
// global into LDS at byte offset `lds`, inserting 4 dwords of padding after
// every 16 dwords (64B data + 16B pad = 80B LDS row = stride-40 bf16 layout).
// D# packing per CDNA5 ISA. This toolchain exposes the
// 6-arg builtin: (g0 u32x4, g1 i32x8, g2 i32x4, g3 i32x4, i32x8, cpol).
// ---------------------------------------------------------------------------
__device__ __forceinline__ void tdm_load_tile_bf16(unsigned lds,
                                                   unsigned long long ga,
                                                   int strideElems) {
  u32x4 g0;
  g0[0] = 1u;                                   // count=1, user mode
  g0[1] = lds;                                  // lds_addr (bytes)
  g0[2] = (unsigned)ga;                         // global_addr[31:0]
  g0[3] = ((unsigned)(ga >> 32) & 0x01FFFFFFu)  // global_addr[56:32]
          | 0x80000000u;                        // type=2 ("image")
  i32x8 g1;
  g1[0] = 0x06D10000;        // data_size=2B | pad_en | pad_interval=16dw | pad_amount=4dw
  g1[1] = (int)0xFFFF0000;   // tensor_dim0 lo16 = 0xFFFF  (huge, tiles in-bounds)
  g1[2] = (int)0xFFFF7FFF;   // tensor_dim0 hi = 0x7FFF | tensor_dim1 lo16 = 0xFFFF
  g1[3] = 0x00207FFF;        // tensor_dim1 hi = 0x7FFF | tile_dim0 = 32
  g1[4] = 128;               // tile_dim1 = 128, tile_dim2 = 0
  g1[5] = strideElems;       // tensor_dim0_stride lo32 (elements)
  g1[6] = 0;                 // stride hi | dim1_stride lo
  g1[7] = 0;
  i32x4 gz; gz[0] = 0; gz[1] = 0; gz[2] = 0; gz[3] = 0;
  i32x8 z8; z8[0]=0; z8[1]=0; z8[2]=0; z8[3]=0; z8[4]=0; z8[5]=0; z8[6]=0; z8[7]=0;
  __builtin_amdgcn_tensor_load_to_lds(g0, g1, gz, gz, z8, 0);
}

// ---------------------------------------------------------------------------
// Fused GEMM + bias + ReLU using V_WMMA_F32_16X16X32_BF16.
// C[b][n] = relu( sum_d A[b][d] * Wg[d][n] + bias[n] ), batched over blockIdx.z.
// Block tile 128x128, K tile 32. 256 threads = 8 waves; wave grid 4(M)x2(N),
// each wave owns a 32x64 sub-tile = 2x4 WMMA tiles.
// LDS: sA row-major [row][k] bf16 (stride 40 = pad), sB transposed [n][k] bf16.
// ABF16 path stages A via the Tensor Data Mover (TENSORcnt).
// ---------------------------------------------------------------------------
template<bool ABF16, bool OF32, bool OBF16>
__global__ __launch_bounds__(256) void gemm_relu_kernel(
    const void* __restrict__ Ap, const float* __restrict__ Wg,
    const float* __restrict__ bias, float* __restrict__ outF,
    __bf16* __restrict__ outB, int Bm, int D, int N)
{
  __shared__ __align__(16) __bf16 sA[128 * 40];
  __shared__ __align__(16) __bf16 sB[128 * 40];

  const int m    = blockIdx.z;
  const int brow = blockIdx.y * 128;
  const int bcol = blockIdx.x * 128;
  const size_t aoff = (size_t)m * Bm * D;
  const size_t woff = (size_t)m * D * N;
  const float* biasm = bias + (size_t)m * N;
  const size_t ooff = (size_t)m * Bm * N;

  const int tid  = threadIdx.x;
  const int lane = tid & 31, wid = tid >> 5;
  const int wm = wid & 3, wn = wid >> 2;   // 4 waves along M, 2 along N
  const int lr = lane & 15, lh = lane >> 4;

  v8f acc[2][4];
  {
    AccU z;
    #pragma unroll
    for (int e = 0; e < 8; ++e) z.f[e] = 0.0f;
    #pragma unroll
    for (int r = 0; r < 2; ++r)
      #pragma unroll
      for (int c = 0; c < 4; ++c) acc[r][c] = z.v;
  }

  // ---- stage B tile transposed: sB[n][k] = W[k0+k][bcol+n] ----
  auto stageB = [&](int k0, bool full) {
    const int tk  = tid >> 5;            // 0..7
    const int tn4 = (tid & 31) * 4;      // 0..124
    #pragma unroll
    for (int p = 0; p < 4; ++p) {
      const int k  = p * 8 + tk;
      const int gk = k0 + k;
      float x0 = 0.f, x1 = 0.f, x2 = 0.f, x3 = 0.f;
      if (full || gk < D) {
        float4 v = *(const float4*)&Wg[woff + (size_t)gk * N + bcol + tn4];
        x0 = v.x; x1 = v.y; x2 = v.z; x3 = v.w;
      }
      sB[(tn4 + 0) * 40 + k] = (__bf16)x0;
      sB[(tn4 + 1) * 40 + k] = (__bf16)x1;
      sB[(tn4 + 2) * 40 + k] = (__bf16)x2;
      sB[(tn4 + 3) * 40 + k] = (__bf16)x3;
    }
    // prefetch next weight K-tile (global_prefetch_b8); clamp keeps it valid
    int nk = k0 + 32 + tk; if (nk >= D) nk = D - 1;
    __builtin_prefetch(&Wg[woff + (size_t)nk * N + bcol + tn4], 0, 0);
  };

  // ---- build fragments from LDS, issue 2x4 WMMAs ----
  // A 16-bit 16x32 layout: lane = (M = lane&15, half = lane>>4);
  //   elems 0..7 -> K = 8*half..8*half+7 ; elems 8..15 -> K = 16+8*half..23+8*half
// B: lane holds column N = lane&15, same K order as A (from sB[n][k])
  auto compute = [&]() {
    FragU af[2], bfr[4];
    #pragma unroll
    for (int rt = 0; rt < 2; ++rt) {
      const __bf16* pa = &sA[(wm * 32 + rt * 16 + lr) * 40 + 8 * lh];
      af[rt].h[0] = *(const v8bf*)pa;
      af[rt].h[1] = *(const v8bf*)(pa + 16);
    }
    #pragma unroll
    for (int ct = 0; ct < 4; ++ct) {
      const __bf16* pb = &sB[(wn * 64 + ct * 16 + lr) * 40 + 8 * lh];
      bfr[ct].h[0] = *(const v8bf*)pb;
      bfr[ct].h[1] = *(const v8bf*)(pb + 16);
    }
    #pragma unroll
    for (int rt = 0; rt < 2; ++rt)
      #pragma unroll
      for (int ct = 0; ct < 4; ++ct)
        acc[rt][ct] = __builtin_amdgcn_wmma_f32_16x16x32_bf16(
            false, af[rt].v, false, bfr[ct].v, (short)0, acc[rt][ct], false, false);
  };

  if constexpr (ABF16) {
    // A is bf16 row-major in workspace; D is a multiple of 32 here.
    const __bf16* Abase = (const __bf16*)Ap + aoff + (size_t)brow * D;
    const unsigned ldsA = (unsigned)(uintptr_t)&sA[0];  // low 32b = LDS offset
    const int KT = D >> 5;
    for (int kt = 0; kt < KT; ++kt) {
      const int k0 = kt * 32;
      if (wid == 0)  // one wave drives the DMA; EXEC-independent issue
        tdm_load_tile_bf16(ldsA, (unsigned long long)(uintptr_t)(Abase + k0), D);
      stageB(k0, true);
      if (wid == 0) __builtin_amdgcn_s_wait_tensorcnt(0);
      __syncthreads();
      compute();
      __syncthreads();
    }
  } else {
    // A is f32 (input x): convert while staging. Peel the K tail (D = 2000).
    const float* Af = (const float*)Ap;
    const int trr = tid >> 3, tk4 = (tid & 7) * 4;
    const int KTfull = D >> 5;
    for (int kt = 0; kt < KTfull; ++kt) {
      const int k0 = kt * 32;
      #pragma unroll
      for (int p = 0; p < 4; ++p) {
        const int row = p * 32 + trr;
        float4 v = *(const float4*)&Af[aoff + (size_t)(brow + row) * D + k0 + tk4];
        __bf16* dst = &sA[row * 40 + tk4];
        dst[0] = (__bf16)v.x; dst[1] = (__bf16)v.y;
        dst[2] = (__bf16)v.z; dst[3] = (__bf16)v.w;
      }
      stageB(k0, true);
      __syncthreads();
      compute();
      __syncthreads();
    }
    if (D & 31) {  // zero-padded K tail, single iteration
      const int k0 = KTfull * 32;
      #pragma unroll
      for (int p = 0; p < 4; ++p) {
        const int row = p * 32 + trr;
        const float* src = &Af[aoff + (size_t)(brow + row) * D + k0 + tk4];
        __bf16* dst = &sA[row * 40 + tk4];
        dst[0] = (__bf16)((k0 + tk4 + 0 < D) ? src[0] : 0.0f);
        dst[1] = (__bf16)((k0 + tk4 + 1 < D) ? src[1] : 0.0f);
        dst[2] = (__bf16)((k0 + tk4 + 2 < D) ? src[2] : 0.0f);
        dst[3] = (__bf16)((k0 + tk4 + 3 < D) ? src[3] : 0.0f);
      }
      stageB(k0, false);
      __syncthreads();
      compute();
      __syncthreads();
    }
  }

  // ---- epilogue: bias + ReLU, write f32 and/or bf16 ----
  // C/D layout: VGPR s -> row = s + 8*(lane>>4), col = lane&15
  #pragma unroll
  for (int rt = 0; rt < 2; ++rt) {
    #pragma unroll
    for (int ct = 0; ct < 4; ++ct) {
      const int colg = bcol + wn * 64 + ct * 16 + lr;
      const float bv = biasm[colg];
      AccU u; u.v = acc[rt][ct];
      #pragma unroll
      for (int s = 0; s < 8; ++s) {
        const int rowg = brow + wm * 32 + rt * 16 + s + 8 * lh;
        const float val = fmaxf(u.f[s] + bv, 0.0f);
        const size_t o = ooff + (size_t)rowg * N + colg;
        if constexpr (OF32)  outF[o] = val;
        if constexpr (OBF16) outB[o] = (__bf16)val;
      }
    }
  }
}

// ---------------------------------------------------------------------------
// Wt[m][b][k][c] = sum_f W[m][k][f] * t[m][b][f][c],  t[m][b][f][c]=z[m][b][2f+c]
// ---------------------------------------------------------------------------
__global__ __launch_bounds__(128) void wt_kernel(
    const float* __restrict__ z, const float* __restrict__ W,
    float* __restrict__ Wt, int Bn)
{
  const int mb = blockIdx.x;              // m*Bn + b
  const int m = mb / Bn;
  __shared__ float sz[256];
  const int t = threadIdx.x;
  ((float2*)sz)[t] = ((const float2*)&z[(size_t)mb * 256])[t];
  __syncthreads();
  if (t < 80) {
    const int k = t >> 1, c = t & 1;
    const float* wr = &W[(size_t)m * 40 * 128 + (size_t)k * 128];
    float acc = 0.0f;
    #pragma unroll 4
    for (int f = 0; f < 128; ++f) acc += wr[f] * sz[2 * f + c];
    Wt[(size_t)mb * 80 + t] = acc;        // layout [k][c], t = 2k+c
  }
}

// ---------------------------------------------------------------------------
// afm[p][b][c][d] = sum_f sum_g t_i[f][c] * aff[aid][f][g] * t_j[g][d]
// sym pairs p: (0,0,a0)(0,1,a0)(0,2,a1)(1,1,a0)(1,2,a0)(2,2,a0)
// ---------------------------------------------------------------------------
__global__ __launch_bounds__(128) void afm_kernel(
    const float* __restrict__ z, const float* __restrict__ aff,
    float* __restrict__ afm, int Bn)
{
  const int pi[6] = {0,0,0,1,1,2};
  const int pj[6] = {0,1,2,1,2,2};
  const int pa[6] = {0,0,1,0,0,0};   // reference's aff_idx bookkeeping
  const int pb = blockIdx.x;
  const int p = pb / Bn, b = pb % Bn;
  const int i = pi[p], j = pj[p], a = pa[p];

  __shared__ float ti[256], tj[256];
  __shared__ float red[4][128];
  const int t = threadIdx.x;
  ((float2*)ti)[t] = ((const float2*)&z[((size_t)i * Bn + b) * 256])[t];
  ((float2*)tj)[t] = ((const float2*)&z[((size_t)j * Bn + b) * 256])[t];
  __syncthreads();

  const int g = t;
  float y0 = 0.0f, y1 = 0.0f;
  const float* am = &aff[(size_t)a * 16384 + g];
  #pragma unroll 4
  for (int f = 0; f < 128; ++f) {
    const float av = am[(size_t)f * 128];   // coalesced across g
    y0 += ti[2 * f] * av;
    y1 += ti[2 * f + 1] * av;
  }
  const float tj0 = tj[2 * g], tj1 = tj[2 * g + 1];
  red[0][t] = y0 * tj0; red[1][t] = y0 * tj1;
  red[2][t] = y1 * tj0; red[3][t] = y1 * tj1;
  __syncthreads();
  for (int s = 64; s > 0; s >>= 1) {
    if (t < s)
      #pragma unroll
      for (int q = 0; q < 4; ++q) red[q][t] += red[q][t + s];
    __syncthreads();
  }
  if (t < 4) afm[((size_t)p * Bn + b) * 4 + t] = red[t][0];
}

// ---------------------------------------------------------------------------
// Per ordered pair (i!=j): H = tanh(Wt[max] + Wt[min] @ Mat), Mat = afm or afm^T,
// s[d] = sum_k Wh[i][j][k]*H[k][d], a = softmax(s), out = a*t[i]
// ---------------------------------------------------------------------------
__global__ __launch_bounds__(128) void head_kernel(
    const float* __restrict__ z, const float* __restrict__ Wt,
    const float* __restrict__ afm, const float* __restrict__ Wh,
    float* __restrict__ outc, int Bn)
{
  const int oi[6] = {0,0,1,1,2,2};
  const int oj[6] = {1,2,0,2,0,1};
  const int ob = blockIdx.x;
  const int op = ob / Bn, b = ob % Bn;
  const int i = oi[op], j = oj[op];
  const int lo = (i < j) ? i : j, hi = (i < j) ? j : i;
  const int s = (lo == 0) ? ((hi == 1) ? 1 : 2) : 4;  // sym index of (lo,hi)
  const bool tr = (i > j);

  __shared__ float wA[80], wM[80], mat[4], red[80], sd[2];
  const int t = threadIdx.x;
  if (t < 80) {
    wA[t] = Wt[((size_t)hi * Bn + b) * 80 + t];
    wM[t] = Wt[((size_t)lo * Bn + b) * 80 + t];
  }
  if (t < 4) mat[t] = afm[((size_t)s * Bn + b) * 4 + t];
  __syncthreads();

  if (t < 80) {
    const int k = t >> 1, d = t & 1;
    const float m0 = tr ? mat[d * 2 + 0] : mat[0 * 2 + d];
    const float m1 = tr ? mat[d * 2 + 1] : mat[1 * 2 + d];
    const float h = tanhf(wA[t] + wM[2 * k] * m0 + wM[2 * k + 1] * m1);
    red[t] = Wh[(size_t)(i * 3 + j) * 40 + k] * h;
  }
  __syncthreads();
  if (t < 2) {
    float acc = 0.0f;
    for (int k = 0; k < 40; ++k) acc += red[2 * k + t];
    sd[t] = acc;
  }
  __syncthreads();
  const float s0 = sd[0], s1 = sd[1];
  const float mx = fmaxf(s0, s1);
  const float e0 = __expf(s0 - mx), e1 = __expf(s1 - mx);
  const float inv = 1.0f / (e0 + e1);
  const float a0 = e0 * inv, a1 = e1 * inv;

  const float2 zt = ((const float2*)&z[((size_t)i * Bn + b) * 256])[t];
  float2 o; o.x = a0 * zt.x; o.y = a1 * zt.y;
  ((float2*)&outc[(size_t)b * 1536 + (size_t)op * 256])[t] = o;
}

// ---------------------------------------------------------------------------
extern "C" void kernel_launch(void* const* d_in, const int* in_sizes, int n_in,
                              void* d_out, int out_size, void* d_ws, size_t ws_size,
                              hipStream_t stream) {
  (void)in_sizes; (void)n_in; (void)out_size; (void)ws_size;
  const float* x   = (const float*)d_in[0];
  const float* w0  = (const float*)d_in[1];
  const float* b0  = (const float*)d_in[2];
  const float* w1  = (const float*)d_in[3];
  const float* b1  = (const float*)d_in[4];
  const float* w2  = (const float*)d_in[5];
  const float* b2  = (const float*)d_in[6];
  const float* aff = (const float*)d_in[7];
  const float* W   = (const float*)d_in[8];
  const float* Wh  = (const float*)d_in[9];
  float* out = (float*)d_out;

  const int M = 3, Bn = 8192, D0 = 2000, H1 = 1024, H2 = 512, Z = 256;

  char* ws = (char*)d_ws;
  __bf16* h1   = (__bf16*)ws;                          // 3*8192*1024 bf16
  __bf16* h2   = (__bf16*)(ws + 50331648);             // 3*8192*512  bf16
  float*  wt   = (float*)(ws + 75497472);              // 3*8192*80   f32
  float*  afmv = (float*)(ws + 83361792);              // 6*8192*4    f32
  float*  zout = out;                                  // (M,B,Z) f32
  float*  cat  = out + (size_t)M * Bn * Z;             // (B, 6*Z) f32

  gemm_relu_kernel<false, false, true><<<dim3(H1 / 128, Bn / 128, M), 256, 0, stream>>>(
      x, w0, b0, nullptr, h1, Bn, D0, H1);
  gemm_relu_kernel<true, false, true><<<dim3(H2 / 128, Bn / 128, M), 256, 0, stream>>>(
      h1, w1, b1, nullptr, h2, Bn, H1, H2);
  gemm_relu_kernel<true, true, false><<<dim3(Z / 128, Bn / 128, M), 256, 0, stream>>>(
      h2, w2, b2, zout, nullptr, Bn, H2, Z);

  wt_kernel<<<M * Bn, 128, 0, stream>>>(zout, W, wt, Bn);
  afm_kernel<<<6 * Bn, 128, 0, stream>>>(zout, aff, afmv, Bn);
  head_kernel<<<6 * Bn, 128, 0, stream>>>(zout, wt, afmv, Wh, cat, Bn);
}
